// FlashAttnModule_64639257805046
// MI455X (gfx1250) — matrix operand order repaired and checked by reading, not compiled, not run on hardware
//
#include <hip/hip_runtime.h>
#include <hip/hip_bf16.h>

// GLM-masked flash-attention forward for MI455X (gfx1250), wave32 WMMA.
// b x s x nh x hs = B x 2048 x 16 x 64, fp32 in/out.
// 256-thread block (8 waves) per (b, h, 128-row q tile); each wave owns
// 16 q rows x 64 dims. KB=64 keys per iteration, staged in LDS as f16.
// Softmax runs in the log2 domain (log2e folded into Q) so exp lowers to a
// bare v_exp_f32; row max uses ds_swizzle xor-reductions; row sum uses the
// matrix unit (P x ones).

typedef __attribute__((ext_vector_type(16))) _Float16 v16h;
typedef __attribute__((ext_vector_type(4)))  _Float16 h4;
typedef __attribute__((ext_vector_type(2)))  _Float16 h2;
typedef __attribute__((ext_vector_type(8)))  float    v8f;
typedef __attribute__((ext_vector_type(4)))  float    f4;
typedef __attribute__((ext_vector_type(2)))  float    f2;

constexpr int   S_LEN   = 2048;
constexpr int   NH      = 16;
constexpr int   HS      = 64;
constexpr int   QB      = 128;   // q rows per block (8 waves x 16)
constexpr int   KBK     = 64;    // keys per iteration
constexpr float NEG_INF = -1e30f;
// (1/sqrt(64)) * log2(e): scores come out in log2 units, exp -> exp2.
constexpr float QSCALE  = 0.125f * 1.44269504088896340736f;

__device__ __forceinline__ float fexp2(float x) {
  return __builtin_amdgcn_exp2f(x);   // v_exp_f32
}

__device__ __forceinline__ v8f wmma16(v16h a, v16h b, v8f c) {
  return __builtin_amdgcn_wmma_f32_16x16x32_f16(false, a, false, b,
                                                (short)0, c, false, false);
}

// Max-reduce across the 16 lanes of a half-wave via ds_swizzle bit-mask mode:
// offset[4:0]=and_mask(0x1F), offset[9:5]=or_mask(0), offset[14:10]=xor_mask.
__device__ __forceinline__ float rmax16(float x) {
  union { float f; int i; } u, s;
  u.f = x;
  s.i = __builtin_amdgcn_ds_swizzle(u.i, 0x041F); u.f = fmaxf(u.f, s.f); // xor 1
  s.i = __builtin_amdgcn_ds_swizzle(u.i, 0x081F); u.f = fmaxf(u.f, s.f); // xor 2
  s.i = __builtin_amdgcn_ds_swizzle(u.i, 0x101F); u.f = fmaxf(u.f, s.f); // xor 4
  s.i = __builtin_amdgcn_ds_swizzle(u.i, 0x201F); u.f = fmaxf(u.f, s.f); // xor 8
  return u.f;
}

__global__ __launch_bounds__(256)
void glm_flash_fwd(const float* __restrict__ Q, const float* __restrict__ K,
                   const float* __restrict__ V, const int* __restrict__ GLM,
                   float* __restrict__ Out) {
  __shared__ __align__(16) _Float16 Ksh[KBK][72];     // [key][dim], padded
  __shared__ __align__(16) _Float16 VshT[HS][72];     // [dim][key], padded
  __shared__ __align__(16) _Float16 Psh[8][16][66];   // per-wave P transpose

  const int qblk = blockIdx.x, h = blockIdx.y, b = blockIdx.z;
  const int Q0   = qblk * QB;
  const int tid  = threadIdx.x;
  const int wave = tid >> 5, lane = tid & 31;
  const int g    = lane >> 4;        // half-wave group (rows 8g..8g+7)
  const int mc   = lane & 15;        // column / row-in-group index
  const int qw0  = Q0 + wave * 16;   // this wave's first q row
  const int bp   = GLM[b];           // GLM break point

  const size_t rowStride = (size_t)NH * HS;
  const float* qB = Q + (size_t)b * S_LEN * rowStride + (size_t)h * HS;
  const float* kB = K + (size_t)b * S_LEN * rowStride + (size_t)h * HS;
  const float* vB = V + (size_t)b * S_LEN * rowStride + (size_t)h * HS;

  // ---- Load Q tile into A-layout f16 registers, pre-scaled ----
  // A 16x32: lane(g,mc) holds row mc, k = (v/4)*16 + g*8 + (v%4)*2 + half.
  v16h Aq0, Aq1;
  {
    const float* qr = qB + (size_t)(qw0 + mc) * rowStride;
#pragma unroll
    for (int v = 0; v < 8; ++v) {
      const int kd = (v >> 2) * 16 + g * 8 + (v & 3) * 2;
      f2 d0 = *(const f2*)(qr + kd);
      f2 d1 = *(const f2*)(qr + 32 + kd);
      Aq0[2 * v]     = (_Float16)(d0.x * QSCALE);
      Aq0[2 * v + 1] = (_Float16)(d0.y * QSCALE);
      Aq1[2 * v]     = (_Float16)(d1.x * QSCALE);
      Aq1[2 * v + 1] = (_Float16)(d1.y * QSCALE);
    }
  }

  // Constant all-ones B operand: rowsum(P) via two WMMAs instead of shuffles.
  v16h Bones;
#pragma unroll
  for (int i = 0; i < 16; ++i) Bones[i] = (_Float16)1.0f;

  v8f Oacc[4] = {v8f{}, v8f{}, v8f{}, v8f{}};
  float m_run[8], l_run[8], cr[8];
#pragma unroll
  for (int r = 0; r < 8; ++r) { m_run[r] = NEG_INF; l_run[r] = 0.0f; }

  // Allowed keys form a prefix: k < max(q_max+1, bp).
  int maxk = Q0 + QB;
  if (bp > maxk) maxk = bp;
  int kb_end = (maxk + KBK - 1) / KBK * KBK;
  if (kb_end > S_LEN) kb_end = S_LEN;
  // This wave needs keys k < kw_end only.
  int kw_end = qw0 + 16;
  if (bp > kw_end) kw_end = bp;

  for (int kb = 0; kb < kb_end; kb += KBK) {
    // ---- Stage K (row-major) and V (transposed) as f16 in LDS ----
#pragma unroll
    for (int it = 0; it < 4; ++it) {
      const int i   = tid + it * 256;   // KBK rows * 16 float4 = 1024 slots
      const int row = i >> 4;
      const int c4  = (i & 15) * 4;
      f4 kd = *(const f4*)(kB + (size_t)(kb + row) * rowStride + c4);
      f4 vd = *(const f4*)(vB + (size_t)(kb + row) * rowStride + c4);
      h4 kh;
      kh.x = (_Float16)kd.x; kh.y = (_Float16)kd.y;
      kh.z = (_Float16)kd.z; kh.w = (_Float16)kd.w;
      *(h4*)&Ksh[row][c4] = kh;              // one b64 store
      VshT[c4 + 0][row] = (_Float16)vd.x;    // scattered b16 stores
      VshT[c4 + 1][row] = (_Float16)vd.y;
      VshT[c4 + 2][row] = (_Float16)vd.z;
      VshT[c4 + 3][row] = (_Float16)vd.w;
    }
    __syncthreads();

    // Prefetch next tile (global_prefetch_b8), one 128B line per thread.
    if (kb + KBK < kb_end) {
      const int line = tid & 127;
      const float* base = (tid < 128) ? kB : vB;
      __builtin_prefetch(
          base + (size_t)(kb + KBK + (line >> 1)) * rowStride + (line & 1) * 32,
          0, 1);
    }

    if (kb < kw_end) {   // wave-uniform: skip fully-masked blocks
      // ---- S = (Q * log2e / sqrt(hs)) * K^T, four 16-key subtiles ----
      v8f S[4];
#pragma unroll
      for (int sub = 0; sub < 4; ++sub) {
        v8f acc = {};
#pragma unroll
        for (int c = 0; c < 2; ++c) {
          v16h Bk;
#pragma unroll
          for (int v = 0; v < 8; ++v) {
            const int kk = c * 32 + (v >> 2) * 16 + g * 8 + (v & 3) * 2;
            h2 p = *(const h2*)&Ksh[sub * 16 + mc][kk];
            Bk[2 * v] = p.x; Bk[2 * v + 1] = p.y;
          }
          acc = wmma16(c == 0 ? Aq0 : Aq1, Bk, acc);
        }
        S[sub] = acc;
      }

      // ---- Mask (only boundary blocks take this path) ----
      const bool fullBlock =
          (kb + KBK <= qw0 + 1) || (kb + KBK <= bp);
      if (!fullBlock) {
        const int qoff = qw0 + 8 * g;   // rows qoff..qoff+7 for this lane
#pragma unroll
        for (int sub = 0; sub < 4; ++sub) {
          const int kc  = kb + sub * 16 + mc;
          const bool pre = kc < bp;
#pragma unroll
          for (int r = 0; r < 8; ++r) {
            if (!(pre || (kc <= qoff + r))) S[sub][r] = NEG_INF;
          }
        }
      }

      // ---- Online softmax (log2 domain): row max, exp2, rescale O ----
#pragma unroll
      for (int r = 0; r < 8; ++r) {
        float ml = fmaxf(fmaxf(S[0][r], S[1][r]), fmaxf(S[2][r], S[3][r]));
        ml = rmax16(ml);
        const float mn = fmaxf(m_run[r], ml);
        cr[r] = fexp2(m_run[r] - mn);
        m_run[r] = mn;
#pragma unroll
        for (int sub = 0; sub < 4; ++sub) S[sub][r] = fexp2(S[sub][r] - mn);
        Oacc[0][r] *= cr[r]; Oacc[1][r] *= cr[r];
        Oacc[2][r] *= cr[r]; Oacc[3][r] *= cr[r];
      }

      // ---- Transpose P (D layout -> A layout) through per-wave LDS ----
#pragma unroll
      for (int r = 0; r < 8; ++r) {
#pragma unroll
        for (int sub = 0; sub < 4; ++sub)
          Psh[wave][8 * g + r][sub * 16 + mc] = (_Float16)S[sub][r];
      }
      asm volatile("s_wait_dscnt 0" ::: "memory");  // per-wave DS RAW fence

      v16h Ap0, Ap1;
#pragma unroll
      for (int v = 0; v < 8; ++v) {
        const int kk = (v >> 2) * 16 + g * 8 + (v & 3) * 2;
        h2 p0 = *(const h2*)&Psh[wave][mc][kk];
        h2 p1 = *(const h2*)&Psh[wave][mc][32 + kk];
        Ap0[2 * v] = p0.x; Ap0[2 * v + 1] = p0.y;
        Ap1[2 * v] = p1.x; Ap1[2 * v + 1] = p1.y;
      }

      // ---- Row sums via matrix unit: lsum = P * ones ----
      v8f ls = wmma16(Ap0, Bones, wmma16(Ap1, Bones, v8f{}));
#pragma unroll
      for (int r = 0; r < 8; ++r) l_run[r] = l_run[r] * cr[r] + ls[r];

      // ---- O += P * V ----
#pragma unroll
      for (int dg = 0; dg < 4; ++dg) {
        v16h Bv0, Bv1;
#pragma unroll
        for (int v = 0; v < 8; ++v) {
          const int key = (v >> 2) * 16 + g * 8 + (v & 3) * 2;
          h2 p0 = *(const h2*)&VshT[dg * 16 + mc][key];
          h2 p1 = *(const h2*)&VshT[dg * 16 + mc][32 + key];
          Bv0[2 * v] = p0.x; Bv0[2 * v + 1] = p0.y;
          Bv1[2 * v] = p1.x; Bv1[2 * v + 1] = p1.y;
        }
        Oacc[dg] = wmma16(Ap1, Bv1, wmma16(Ap0, Bv0, Oacc[dg]));
      }
    }
    __syncthreads();   // all LDS reads done before next stage overwrites
  }

  // ---- Epilogue: normalize by l and write out[b, q, h, d] ----
  float* outB = Out + (size_t)b * S_LEN * rowStride + (size_t)h * HS;
#pragma unroll
  for (int r = 0; r < 8; ++r) {
    const int qrow = qw0 + 8 * g + r;
    const float inv = 1.0f / l_run[r];
    float* orow = outB + (size_t)qrow * rowStride;
#pragma unroll
    for (int dg = 0; dg < 4; ++dg)
      orow[dg * 16 + mc] = Oacc[dg][r] * inv;
  }
}

extern "C" void kernel_launch(void* const* d_in, const int* in_sizes, int n_in,
                              void* d_out, int out_size, void* d_ws, size_t ws_size,
                              hipStream_t stream) {
  (void)n_in; (void)out_size; (void)d_ws; (void)ws_size;
  const float* q   = (const float*)d_in[0];
  const float* k   = (const float*)d_in[1];
  const float* v   = (const float*)d_in[2];
  const int*   glm = (const int*)d_in[3];
  float* out = (float*)d_out;
  const int b = in_sizes[3];   // glm_mask has one entry per batch
  dim3 grid(S_LEN / QB, NH, b);
  glm_flash_fwd<<<grid, dim3(256), 0, stream>>>(q, k, v, glm, out);
}
